// ImageTriplaneGenerator_15144054686227
// MI455X (gfx1250) — compile-verified, matrix operand order checked by reading, not run on hardware
//
#include <hip/hip_runtime.h>

// Problem constants (from reference): B=2, Nv=8, C=64, H=W=256, G=128
#define BB    2
#define NV    8
#define BVN   16
#define CCH   64
#define WW    256
#define NPIX  65536           // H*W
#define GRD   128
#define GG2   16384           // G*G

typedef __attribute__((ext_vector_type(8))) int v8i;

// Workspace layout (bytes)
#define WS_BOUNDS   0                          // 6 x u32 encoded (min xyz, max xyz)
#define WS_LOHI     64                         // 6 x f32 (lo0..2, hi0..2), padded
#define WS_VIEWMAT  128                        // 16 views x 12 f32 (P=R*Kinv, t)
#define WS_COUNTS   1024                       // 16*3*16384 u32   (3.0 MB)
#define WS_ACC      (1024 + 16*3*GG2*4)        // 2*3*16384*64 f32 (24 MB)
#define ZERO_WORDS  (16*3*GG2 + BB*3*GG2*CCH)  // counts + acc, contiguous

// ---- order-preserving float<->uint encoding for atomic min/max ----
__device__ __forceinline__ unsigned encf(float f) {
  unsigned u = __float_as_uint(f);
  return (u & 0x80000000u) ? ~u : (u | 0x80000000u);
}
__device__ __forceinline__ float decf(unsigned e) {
  unsigned u = (e & 0x80000000u) ? (e & 0x7FFFFFFFu) : ~e;
  return __uint_as_float(u);
}

// world = d * (P * [u,v,1]) + t   (P = R * Kinv precomputed per view)
__device__ __forceinline__ void projPoint(const float* __restrict__ vm,
                                          float u, float v, float d,
                                          float& wx, float& wy, float& wz) {
  float px = fmaf(vm[0], u, fmaf(vm[1], v, vm[2]));
  float py = fmaf(vm[3], u, fmaf(vm[4], v, vm[5]));
  float pz = fmaf(vm[6], u, fmaf(vm[7], v, vm[8]));
  wx = fmaf(d, px, vm[9]);
  wy = fmaf(d, py, vm[10]);
  wz = fmaf(d, pz, vm[11]);
}

__device__ __forceinline__ int cellOf(float nrm) {
  float c = fminf(fmaxf((nrm * 0.5f + 0.5f) * (float)(GRD - 1), 0.0f),
                  (float)(GRD - 1));
  return (int)c;  // truncation == floor for non-negative
}

// ---- kernel 1: zero counts+acc, init bounds, build per-view P,t ----
__global__ void tri_init(float* __restrict__ ws_f, unsigned* __restrict__ ws_u,
                         const float* __restrict__ c2w,
                         const float* __restrict__ K) {
  unsigned gid = blockIdx.x * blockDim.x + threadIdx.x;
  unsigned stride = gridDim.x * blockDim.x;
  unsigned* zbase = ws_u + WS_COUNTS / 4;
  for (unsigned i = gid; i < (unsigned)ZERO_WORDS; i += stride) zbase[i] = 0u;

  if (gid < 3) {
    ws_u[WS_BOUNDS / 4 + gid]     = encf(1e30f);   // running min
    ws_u[WS_BOUNDS / 4 + 3 + gid] = encf(-1e30f);  // running max
  }
  if (gid < BVN) {
    const float* k = K + gid * 9;
    float k0 = k[0], k1 = k[1], k2 = k[2];
    float k3 = k[3], k4 = k[4], k5 = k[5];
    float k6 = k[6], k7 = k[7], k8 = k[8];
    float det = k0 * (k4 * k8 - k5 * k7) - k1 * (k3 * k8 - k5 * k6) +
                k2 * (k3 * k7 - k4 * k6);
    float id = 1.0f / det;
    float ki[9];
    ki[0] = (k4 * k8 - k5 * k7) * id; ki[1] = (k2 * k7 - k1 * k8) * id; ki[2] = (k1 * k5 - k2 * k4) * id;
    ki[3] = (k5 * k6 - k3 * k8) * id; ki[4] = (k0 * k8 - k2 * k6) * id; ki[5] = (k2 * k3 - k0 * k5) * id;
    ki[6] = (k3 * k7 - k4 * k6) * id; ki[7] = (k1 * k6 - k0 * k7) * id; ki[8] = (k0 * k4 - k1 * k3) * id;
    const float* m = c2w + gid * 16;  // 4x4 row-major
    float* vm = ws_f + WS_VIEWMAT / 4 + gid * 12;
    for (int j = 0; j < 3; ++j) {
      float r0 = m[j * 4 + 0], r1 = m[j * 4 + 1], r2 = m[j * 4 + 2];
      vm[j * 3 + 0] = r0 * ki[0] + r1 * ki[3] + r2 * ki[6];
      vm[j * 3 + 1] = r0 * ki[1] + r1 * ki[4] + r2 * ki[7];
      vm[j * 3 + 2] = r0 * ki[2] + r1 * ki[5] + r2 * ki[8];
      vm[9 + j] = m[j * 4 + 3];  // t
    }
  }
}

// ---- kernel 2: scene bounds (projection pass 1, +0.01) ----
__global__ void tri_bounds(const float* __restrict__ depths,
                           const float* __restrict__ ws_f,
                           unsigned* __restrict__ ws_u) {
  __shared__ unsigned smin[3], smax[3];
  if (threadIdx.x < 3) { smin[threadIdx.x] = 0xFFFFFFFFu; smax[threadIdx.x] = 0u; }
  __syncthreads();

  unsigned gid = blockIdx.x * blockDim.x + threadIdx.x;
  unsigned bv = gid >> 16, n = gid & 0xFFFFu;
  float d0 = depths[gid];
  if (d0 != 0.0f) {
    const float* vm = ws_f + WS_VIEWMAT / 4 + bv * 12;
    float u = (float)(n & (WW - 1)), v = (float)(n >> 8);
    float wx, wy, wz;
    projPoint(vm, u, v, d0 + 0.01f, wx, wy, wz);
    atomicMin(&smin[0], encf(wx)); atomicMax(&smax[0], encf(wx));
    atomicMin(&smin[1], encf(wy)); atomicMax(&smax[1], encf(wy));
    atomicMin(&smin[2], encf(wz)); atomicMax(&smax[2], encf(wz));
  }
  __syncthreads();
  if (threadIdx.x < 3) {
    atomicMin(&ws_u[WS_BOUNDS / 4 + threadIdx.x], smin[threadIdx.x]);
    atomicMax(&ws_u[WS_BOUNDS / 4 + 3 + threadIdx.x], smax[threadIdx.x]);
  }
}

// ---- kernel 3: finalize bounds, emit scene_bounds output ----
__global__ void tri_bfin(float* __restrict__ ws_f,
                         const unsigned* __restrict__ ws_u,
                         float* __restrict__ out) {
  if (blockIdx.x == 0 && threadIdx.x == 0) {
    for (int i = 0; i < 3; ++i) {
      float lo = fminf(decf(ws_u[i]), 0.0f);
      float hi = fmaxf(decf(ws_u[3 + i]), 0.0f);
      float ext = hi - lo;
      float l = lo - 0.05f * ext, h = hi + 0.05f * ext;
      ws_f[WS_LOHI / 4 + i] = l;
      ws_f[WS_LOHI / 4 + 3 + i] = h;
      out[BB * 3 * CCH * GG2 + 2 * i]     = l;  // (lo0,hi0,lo1,hi1,lo2,hi2)
      out[BB * 3 * CCH * GG2 + 2 * i + 1] = h;
    }
  }
}

// ---- kernel 4: per-(view,plane,cell) point counts (pass 2, +0.02) ----
__global__ void tri_count(const float* __restrict__ depths,
                          const float* __restrict__ ws_f,
                          unsigned* __restrict__ counts) {
  unsigned gid = blockIdx.x * blockDim.x + threadIdx.x;
  unsigned bv = gid >> 16, n = gid & 0xFFFFu;
  float d0 = depths[gid];
  if (d0 == 0.0f) return;
  const float* vm = ws_f + WS_VIEWMAT / 4 + bv * 12;
  float u = (float)(n & (WW - 1)), v = (float)(n >> 8);
  float wx, wy, wz;
  projPoint(vm, u, v, d0 + 0.02f, wx, wy, wz);
  const float* lh = ws_f + WS_LOHI / 4;
  float nx = 2.0f * (wx - lh[0]) / (lh[3] - lh[0]) - 1.0f;
  float ny = 2.0f * (wy - lh[1]) / (lh[4] - lh[1]) - 1.0f;
  float nz = 2.0f * (wz - lh[2]) / (lh[5] - lh[2]) - 1.0f;
  int cx = cellOf(nx), cy = cellOf(ny), cz = cellOf(nz);
  atomicAdd(&counts[(bv * 3 + 0) * GG2 + cx * GRD + cy], 1u);  // plane (0,1)
  atomicAdd(&counts[(bv * 3 + 1) * GG2 + cx * GRD + cz], 1u);  // plane (0,2)
  atomicAdd(&counts[(bv * 3 + 2) * GG2 + cy * GRD + cz], 1u);  // plane (1,2)
}

// ---- kernel 5: weighted feature scatter (L2-resident fp32 atomics) ----
__global__ void tri_scatter(const float* __restrict__ depths,
                            const float* __restrict__ feat,
                            const float* __restrict__ ws_f,
                            const unsigned* __restrict__ counts,
                            float* __restrict__ acc) {
  unsigned gid = blockIdx.x * blockDim.x + threadIdx.x;
  unsigned bv = gid >> 16, n = gid & 0xFFFFu;
  float d0 = depths[gid];
  if (d0 == 0.0f) return;
  const float* vm = ws_f + WS_VIEWMAT / 4 + bv * 12;
  float u = (float)(n & (WW - 1)), v = (float)(n >> 8);
  float wx, wy, wz;
  projPoint(vm, u, v, d0 + 0.02f, wx, wy, wz);
  const float* lh = ws_f + WS_LOHI / 4;
  float nx = 2.0f * (wx - lh[0]) / (lh[3] - lh[0]) - 1.0f;
  float ny = 2.0f * (wy - lh[1]) / (lh[4] - lh[1]) - 1.0f;
  float nz = 2.0f * (wz - lh[2]) / (lh[5] - lh[2]) - 1.0f;
  int cx = cellOf(nx), cy = cellOf(ny), cz = cellOf(nz);
  int c01 = cx * GRD + cy, c02 = cx * GRD + cz, c12 = cy * GRD + cz;
  // weight = 1/max(cnt,1): turns per-view segment-mean + cross-view sum into
  // a single scaled scatter (avoids per-view accumulators)
  float w0 = 1.0f / fmaxf((float)counts[(bv * 3 + 0) * GG2 + c01], 1.0f);
  float w1 = 1.0f / fmaxf((float)counts[(bv * 3 + 1) * GG2 + c02], 1.0f);
  float w2 = 1.0f / fmaxf((float)counts[(bv * 3 + 2) * GG2 + c12], 1.0f);
  unsigned b = bv >> 3;
  float* a0 = acc + ((b * 3 + 0) * GG2 + c01) * CCH;
  float* a1 = acc + ((b * 3 + 1) * GG2 + c02) * CCH;
  float* a2 = acc + ((b * 3 + 2) * GG2 + c12) * CCH;
  unsigned fbase = bv * CCH * NPIX + n;  // channels strided by NPIX, lanes coalesced
  for (int c = 0; c < CCH; ++c) {
    if (c + 8 < CCH)  // CDNA5 global_prefetch_b8 for the streaming channel walk
      __builtin_prefetch(&feat[fbase + (unsigned)(c + 8) * NPIX], 0, 0);
    float f = feat[fbase + (unsigned)c * NPIX];
    __hip_atomic_fetch_add(a0 + c, f * w0, __ATOMIC_RELAXED, __HIP_MEMORY_SCOPE_AGENT);
    __hip_atomic_fetch_add(a1 + c, f * w1, __ATOMIC_RELAXED, __HIP_MEMORY_SCOPE_AGENT);
    __hip_atomic_fetch_add(a2 + c, f * w2, __ATOMIC_RELAXED, __HIP_MEMORY_SCOPE_AGENT);
  }
}

// ---- kernel 6: finalize. po = sum over 8 views of (cnt>0) via
//      V_WMMA_I32_16X16X64_IU8 (A = ones over K=0..7, B = occupancy bytes),
//      then tri = acc / max(po, 1e-6). One 32-lane wave per 16-cell block.
__global__ void tri_finalize(const unsigned* __restrict__ counts,
                             const float* __restrict__ acc,
                             float* __restrict__ out) {
  unsigned blk  = blockIdx.x;
  unsigned bp   = blk >> 10;            // 0..5  = b*3+p
  unsigned base = (blk & 1023u) << 4;   // cell block of 16
  unsigned b    = bp / 3, p = bp % 3;
  unsigned lane = threadIdx.x;
  unsigned cell = base + (lane & 15u);
  unsigned half = lane >> 4;

  // B (64x16 iu8): lanes 0-15 carry K=0..15 in V0..V3 (4 bytes/VGPR);
  // views occupy K=0..7 -> V0 = occ(v0..3), V1 = occ(v4..7). lanes 16-31 zero.
  v8i Bv = {0, 0, 0, 0, 0, 0, 0, 0};
  if (half == 0) {
    unsigned o0 = 0, o1 = 0;
    for (int vw = 0; vw < 4; ++vw) {
      unsigned bv0 = b * NV + (unsigned)vw;
      unsigned bv1 = b * NV + 4u + (unsigned)vw;
      o0 |= (counts[(bv0 * 3 + p) * GG2 + cell] ? 1u : 0u) << (8 * vw);
      o1 |= (counts[(bv1 * 3 + p) * GG2 + cell] ? 1u : 0u) << (8 * vw);
    }
    Bv[0] = (int)o0;
    Bv[1] = (int)o1;
  }
  // A (16x64 iu8) = 1 for K=0..7, all rows: lanes 0-15 V0=K0-3, V1=K4-7.
  v8i Av = {0, 0, 0, 0, 0, 0, 0, 0};
  if (half == 0) { Av[0] = 0x01010101; Av[1] = 0x01010101; }
  v8i Cv = {0, 0, 0, 0, 0, 0, 0, 0};

  v8i Dv = __builtin_amdgcn_wmma_i32_16x16x64_iu8(false, Av, false, Bv, Cv,
                                                  false, false);
  // D[m][n] identical over m; VGPR0 lane L holds column L%16 -> this cell.
  float inv = 1.0f / fmaxf((float)Dv[0], 1e-6f);

  const float* ap = acc + ((size_t)bp * GG2 + cell) * CCH;
  float* op = out + (size_t)bp * CCH * GG2 + cell;  // out[(bp*C + c)*GG2 + cell]
  int c0 = (int)half * 32;
  for (int c = c0; c < c0 + 32; ++c) {
    op[(size_t)c * GG2] = ap[c] * inv;
  }
}

extern "C" void kernel_launch(void* const* d_in, const int* in_sizes, int n_in,
                              void* d_out, int out_size, void* d_ws,
                              size_t ws_size, hipStream_t stream) {
  const float* feat   = (const float*)d_in[0];  // (2,8,64,256,256)
  const float* depths = (const float*)d_in[1];  // (2,8,1,256,256)
  const float* c2w    = (const float*)d_in[2];  // (2,8,4,4)
  const float* K      = (const float*)d_in[3];  // (2,8,3,3)
  float* out = (float*)d_out;                   // 2*3*64*128*128 tri + 6 bounds

  float* ws_f = (float*)d_ws;
  unsigned* ws_u = (unsigned*)d_ws;
  unsigned* counts = (unsigned*)((char*)d_ws + WS_COUNTS);
  float* acc = (float*)((char*)d_ws + WS_ACC);

  const int PTS_BLOCKS = (BVN * NPIX) / 256;  // 4096

  tri_init<<<PTS_BLOCKS, 256, 0, stream>>>(ws_f, ws_u, c2w, K);
  tri_bounds<<<PTS_BLOCKS, 256, 0, stream>>>(depths, ws_f, ws_u);
  tri_bfin<<<1, 1, 0, stream>>>(ws_f, ws_u, out);
  tri_count<<<PTS_BLOCKS, 256, 0, stream>>>(depths, ws_f, counts);
  tri_scatter<<<PTS_BLOCKS, 256, 0, stream>>>(depths, feat, ws_f, counts, acc);
  tri_finalize<<<BB * 3 * (GG2 / 16), 32, 0, stream>>>(counts, acc, out);
}
